// HHConv2d_1x1_81827716923772
// MI455X (gfx1250) — matrix operand order repaired and checked by reading, not compiled, not run on hardware
//
#include <hip/hip_runtime.h>
#include <hip/hip_bf16.h>

typedef __attribute__((ext_vector_type(16))) __bf16 v16bf;
typedef __attribute__((ext_vector_type(8)))  float  v8f;

#define CFEAT 256
#define HW    4096
#define NSLAB 32
#define KPAD  264   // LDS row stride in bf16 elements (528B, 16B-aligned, bank-spread)

// ---- bf16 helpers (RNE round, raw-bit storage as unsigned short) ----
__device__ __forceinline__ unsigned short f2bf(float f) {
    union { float f; unsigned u; } uu; uu.f = f;
    unsigned r = uu.u + 0x7FFFu + ((uu.u >> 16) & 1u);
    return (unsigned short)(r >> 16);
}
__device__ __forceinline__ float bf2f(unsigned short h) {
    union { unsigned u; float f; } uu; uu.u = ((unsigned)h) << 16;
    return uu.f;
}

union V16 { v16bf v; uint4 q[2]; };

// -----------------------------------------------------------------------------
// Kernel 1: Householder cascade. One wave32 per output row of W.
// Row r of Q evolves independently: q <- q - (2*(q.v)/(v.v+eps)) * v.
// Each lane owns 8 consecutive columns; two shfl_xor tree reductions per step.
// Emits W split into bf16 hi/lo planes in workspace.
// -----------------------------------------------------------------------------
__global__ __launch_bounds__(256) void hh_cascade_kernel(
    const float* __restrict__ V, unsigned short* __restrict__ Whi,
    unsigned short* __restrict__ Wlo) {
    const int lane = threadIdx.x & 31;
    const int row  = blockIdx.x * 8 + (threadIdx.x >> 5); // 32 blocks * 8 waves = 256 rows
    const int c0   = lane * 8;

    float q[8];
#pragma unroll
    for (int j = 0; j < 8; ++j) q[j] = (c0 + j == row) ? 1.0f : 0.0f;

    for (int i = 0; i < CFEAT - 1; ++i) {
        const float4 a  = *(const float4*)(V + i * CFEAT + c0);
        const float4 b4 = *(const float4*)(V + i * CFEAT + c0 + 4);
        float vv[8] = {a.x, a.y, a.z, a.w, b4.x, b4.y, b4.z, b4.w};
        const int lim = (CFEAT - 2) - i; // columns < lim are constant-1 padding
        float p0 = 0.0f, p1 = 0.0f;
#pragma unroll
        for (int j = 0; j < 8; ++j) {
            float vc = (c0 + j >= lim) ? vv[j] : 1.0f;
            vv[j] = vc;
            p0 += vc * vc;
            p1 += q[j] * vc;
        }
#pragma unroll
        for (int off = 16; off > 0; off >>= 1) {
            p0 += __shfl_xor(p0, off, 32);
            p1 += __shfl_xor(p1, off, 32);
        }
        const float t = 2.0f * p1 / (p0 + 1e-16f);
#pragma unroll
        for (int j = 0; j < 8; ++j) q[j] -= t * vv[j];
    }

    // split into bf16 hi/lo and store 8 contiguous elements (16B) per lane
    union { unsigned short h[8]; uint4 u; } hi, lo;
#pragma unroll
    for (int j = 0; j < 8; ++j) {
        unsigned short hb = f2bf(q[j]);
        hi.h[j] = hb;
        lo.h[j] = f2bf(q[j] - bf2f(hb));
    }
    *(uint4*)(Whi + row * CFEAT + c0) = hi.u;
    *(uint4*)(Wlo + row * CFEAT + c0) = lo.u;
}

// -----------------------------------------------------------------------------
// Kernel 2: y = W @ x per batch via split-bf16 WMMA (hi*hi + hi*lo + lo*hi).
// Block = 256 threads = 8 waves; owns (batch b, 32 spatial columns).
// x tile staged in LDS once as bf16 hi/lo, layout [n][k] (padded stride).
// Wave w computes o-tiles at rows w*32 and w*32+16, n-subtiles {0..15},{16..31}.
// -----------------------------------------------------------------------------
__global__ __launch_bounds__(256) void hh_gemm_kernel(
    const float* __restrict__ x, const unsigned short* __restrict__ Whi,
    const unsigned short* __restrict__ Wlo, float* __restrict__ y) {
    __shared__ unsigned short sh_hi[NSLAB][KPAD];
    __shared__ unsigned short sh_lo[NSLAB][KPAD];

    const int b    = blockIdx.x >> 7;          // 32 batches
    const int slab = blockIdx.x & 127;         // 128 slabs of 32 columns
    const int pos0 = slab * NSLAB;
    const int tid  = threadIdx.x;

    // ---- stage x tile: fp32 -> bf16 hi/lo, transposed to [n][k] ----
    {
        const int n  = tid & 31;
        const int cg = tid >> 5; // wave id = channel group
        const float* xb = x + (size_t)b * CFEAT * HW + pos0 + n;
#pragma unroll 4
        for (int cc = 0; cc < 32; ++cc) {
            const int c = cg * 32 + cc;
            const float xv = xb[(size_t)c * HW];    // coalesced across lanes (n)
            const unsigned short hb = f2bf(xv);
            sh_hi[n][c] = hb;
            sh_lo[n][c] = f2bf(xv - bf2f(hb));
        }
    }
    __syncthreads();

    // ---- compute ----
    const int wave = tid >> 5;
    const int lane = tid & 31;
    const int lm   = lane & 15;
    const int lh   = lane >> 4;
    const int o0   = wave * 32;

    v8f acc[2][2] = {{{0}}}; // [otile][nsub]

    for (int ks = 0; ks < 8; ++ks) {
        const int kb = ks * 32;
        // A fragments: lane holds K = kb + lh*8 + {0..7} and kb + 16 + lh*8 + {0..7}
        V16 Ah[2], Al[2];
#pragma unroll
        for (int ot = 0; ot < 2; ++ot) {
            const int m = o0 + ot * 16 + lm;
            const unsigned short* ph = Whi + (size_t)m * CFEAT + kb + lh * 8;
            const unsigned short* pl = Wlo + (size_t)m * CFEAT + kb + lh * 8;
            Ah[ot].q[0] = *(const uint4*)(ph);
            Ah[ot].q[1] = *(const uint4*)(ph + 16);
            Al[ot].q[0] = *(const uint4*)(pl);
            Al[ot].q[1] = *(const uint4*)(pl + 16);
        }
#pragma unroll
        for (int ns = 0; ns < 2; ++ns) {
            // B fragment: col n = lm, lane holds K = kb + lh*8 + {0..7} and kb + 16 + lh*8 + {0..7}
            const int n = ns * 16 + lm;
            V16 Bh, Bl;
            const unsigned short* sb = &sh_hi[n][kb + lh * 8];
            Bh.q[0] = *(const uint4*)(sb);
            Bh.q[1] = *(const uint4*)(sb + 16);
            const unsigned short* sl = &sh_lo[n][kb + lh * 8];
            Bl.q[0] = *(const uint4*)(sl);
            Bl.q[1] = *(const uint4*)(sl + 16);
#pragma unroll
            for (int ot = 0; ot < 2; ++ot) {
                acc[ot][ns] = __builtin_amdgcn_wmma_f32_16x16x32_bf16(
                    false, Ah[ot].v, false, Bh.v, (short)0, acc[ot][ns], false, false);
                acc[ot][ns] = __builtin_amdgcn_wmma_f32_16x16x32_bf16(
                    false, Ah[ot].v, false, Bl.v, (short)0, acc[ot][ns], false, false);
                acc[ot][ns] = __builtin_amdgcn_wmma_f32_16x16x32_bf16(
                    false, Al[ot].v, false, Bh.v, (short)0, acc[ot][ns], false, false);
            }
        }
    }

    // ---- store: C/D layout VGPR g -> M = g + 8*lh, N = lm ----
    float* yb = y + (size_t)b * CFEAT * HW + pos0;
#pragma unroll
    for (int ot = 0; ot < 2; ++ot) {
#pragma unroll
        for (int ns = 0; ns < 2; ++ns) {
#pragma unroll
            for (int g = 0; g < 8; ++g) {
                const int orow = o0 + ot * 16 + 8 * lh + g;
                yb[(size_t)orow * HW + ns * 16 + lm] = acc[ot][ns][g];
            }
        }
    }
}

// -----------------------------------------------------------------------------
// Kernel 3: pass-through tail outputs (log_det_jac + 0, z)
// -----------------------------------------------------------------------------
__global__ void hh_tail_kernel(const float* __restrict__ ldj,
                               const float* __restrict__ z,
                               float* __restrict__ out) {
    const int t = threadIdx.x;
    const size_t ybase = (size_t)32 * CFEAT * HW; // 33554432
    if (t < 32)  out[ybase + t]       = ldj[t] + 0.0f;
    if (t < 512) out[ybase + 32 + t]  = z[t];
}

extern "C" void kernel_launch(void* const* d_in, const int* in_sizes, int n_in,
                              void* d_out, int out_size, void* d_ws, size_t ws_size,
                              hipStream_t stream) {
    const float* x   = (const float*)d_in[0]; // (32,256,64,64)
    const float* ldj = (const float*)d_in[1]; // (32,)
    const float* z   = (const float*)d_in[2]; // (32,16)
    const float* V   = (const float*)d_in[3]; // (255,256)
    float* out = (float*)d_out;

    unsigned short* Whi = (unsigned short*)d_ws;          // 256*256 bf16 = 128KB
    unsigned short* Wlo = Whi + CFEAT * CFEAT;            // +128KB

    (void)in_sizes; (void)n_in; (void)out_size; (void)ws_size;

    hh_cascade_kernel<<<32, 256, 0, stream>>>(V, Whi, Wlo);
    hh_gemm_kernel<<<32 * 128, 256, 0, stream>>>(x, Whi, Wlo, out);
    hh_tail_kernel<<<1, 512, 0, stream>>>(ldj, z, out);
}
